// RNNRecurrentAutoencoder_38010460570438
// MI455X (gfx1250) — matrix operand order repaired and checked by reading, not compiled, not run on hardware
//
#include <hip/hip_runtime.h>
#include <math.h>

// ---------------- problem constants ----------------
#define SEQ   16384
#define EMB   512
#define HID   1024
#define NWG   8           // persistent workgroups
#define TPB   256         // 8 waves of 32

// ---------------- CDNA5 WMMA types ----------------
typedef __attribute__((ext_vector_type(16))) __bf16 v16bf;
typedef __attribute__((ext_vector_type(8)))  float  v8f;

// ---------------- workspace layout (bytes) ----------------
#define WENC_OFF   0ull
#define WENC_SZ    (512ull * 512ull * 2ull)              // enc Whh bf16 fragments
#define WDEC_OFF   (WENC_OFF + WENC_SZ)
#define WDEC_SZ    (1024ull * 1024ull * 2ull)            // dec Whh bf16 fragments
#define HENC_OFF   (WDEC_OFF + WDEC_SZ)                  // 2 x 512 bf16 (parity)
#define HENC_SZ    (2ull * 1024ull)
#define HDEC_OFF   (HENC_OFF + HENC_SZ)                  // 2 x 1024 bf16 (parity)
#define HDEC_SZ    (2ull * 2048ull)
#define EMBF_OFF   (HDEC_OFF + HDEC_SZ)                  // 512 f32 final encoder state
#define EMBF_SZ    (2048ull)
#define BAR_OFF    (EMBF_OFF + EMBF_SZ)                  // 8 step flags, 64B apart
#define BAR_SZ     (1024ull)
#define PART_OFF   (BAR_OFF + BAR_SZ)                    // SEQ x NWG f32 partial dots

// ---------------- helpers ----------------
__device__ __forceinline__ unsigned short f32_to_bf16_bits(float f) {
    unsigned u = __float_as_uint(f);
    u += 0x7fffu + ((u >> 16) & 1u);        // round-to-nearest-even
    return (unsigned short)(u >> 16);
}
__device__ __forceinline__ unsigned long long pack4_bf16(float a, float b, float c, float d) {
    return (unsigned long long)f32_to_bf16_bits(a)
         | ((unsigned long long)f32_to_bf16_bits(b) << 16)
         | ((unsigned long long)f32_to_bf16_bits(c) << 32)
         | ((unsigned long long)f32_to_bf16_bits(d) << 48);
}
__device__ __forceinline__ unsigned long long pack2_f32(float a, float b) {
    return (unsigned long long)__float_as_uint(a)
         | ((unsigned long long)__float_as_uint(b) << 32);
}
__device__ __forceinline__ void poll_ge(const unsigned* f, unsigned target) {
    while (__hip_atomic_load(f, __ATOMIC_ACQUIRE, __HIP_MEMORY_SCOPE_AGENT) < target)
        __builtin_amdgcn_s_sleep(1);
}
// Async broadcast reload: global (L2, device scope) -> LDS, one qword per lane.
__device__ __forceinline__ void async_qword_to_lds(unsigned lds_addr,
                                                   const unsigned long long* gsrc) {
    asm volatile("global_load_async_to_lds_b64 %0, %1, off scope:SCOPE_DEV"
                 :
                 : "v"(lds_addr), "v"((unsigned long long)gsrc)
                 : "memory");
}
__device__ __forceinline__ void wait_asynccnt0() {
    asm volatile("s_wait_asynccnt 0x0" ::: "memory");
}

// ---------------- prep: f32 -> bf16, swizzled into WMMA A-fragment layout ----------------
// A 16x32 bf16 fragment (contiguous k order used here, matched by the B loads): element (m,k):
//   lane = m + 16*(k>>4) ; half = k&15
// Fragment stored lane-major: frag*1024 + lane*32 + half*2 bytes.
__global__ void prep_kernel(const float* __restrict__ encW,
                            const float* __restrict__ decW,
                            unsigned char* __restrict__ ws) {
    unsigned n = blockIdx.x * blockDim.x + threadIdx.x;
    if (n < 8u) *(unsigned*)(ws + BAR_OFF + n * 64u) = 0u;   // reset step flags
    if (n < 512u * 512u) {                       // encoder Whh (512x512)
        unsigned i = n >> 9, j = n & 511u;
        unsigned wg = i >> 6, rb = (i >> 4) & 3u, m = i & 15u;
        unsigned kb = j >> 5, k = j & 31u;
        unsigned lane = m + (k >> 4) * 16u;
        unsigned half = k & 15u;
        unsigned frag = (wg * 4u + rb) * 16u + kb;
        *(unsigned short*)(ws + WENC_OFF + (unsigned long long)frag * 1024ull +
                           lane * 32u + half * 2u) = f32_to_bf16_bits(encW[i * 512u + j]);
    }
    if (n < 1024u * 1024u) {                     // decoder Whh (1024x1024)
        unsigned i = n >> 10, j = n & 1023u;
        unsigned wg = i >> 7, rb = (i >> 4) & 7u, m = i & 15u;
        unsigned kb = j >> 5, k = j & 31u;
        unsigned lane = m + (k >> 4) * 16u;
        unsigned half = k & 15u;
        unsigned frag = (wg * 8u + rb) * 32u + kb;
        *(unsigned short*)(ws + WDEC_OFF + (unsigned long long)frag * 1024ull +
                           lane * 32u + half * 2u) = f32_to_bf16_bits(decW[i * 1024u + j]);
    }
}

// ---------------- persistent recurrence kernel (8 WGs x 256 threads) ----------------
__global__ void __launch_bounds__(TPB)
rnn_kernel(const float* __restrict__ x,
           const float* __restrict__ enc_Wih, const float* __restrict__ enc_bih,
           const float* __restrict__ enc_bhh,
           const float* __restrict__ dec_Wih, const float* __restrict__ dec_bih,
           const float* __restrict__ dec_bhh,
           const float* __restrict__ out_W,
           unsigned char* __restrict__ ws) {
    __shared__ __align__(32) unsigned short h_bf[HID];   // current hidden state, bf16
    __shared__ __align__(16) float cbuf[128];            // per-row constant term
    __shared__ __align__(16) float owbuf[128];           // out_W slice (dec) / bias (enc)
    __shared__ __align__(16) float ebuf[EMB];            // embedding (decoder init)
    __shared__ __align__(16) float pbuf[16];             // per-writer-lane partials

    const unsigned tid  = threadIdx.x;
    const unsigned wave = tid >> 5;
    const unsigned lane = tid & 31u;
    const unsigned hi   = lane >> 4;                     // 0: rows 0-7, 1: rows 8-15
    const unsigned wg   = blockIdx.x;

    unsigned* flags = (unsigned*)(ws + BAR_OFF);         // flag[w] at w*16 uints
    unsigned long long* g_henc = (unsigned long long*)(ws + HENC_OFF);
    unsigned long long* g_hdec = (unsigned long long*)(ws + HDEC_OFF);
    unsigned long long* g_emb  = (unsigned long long*)(ws + EMBF_OFF);
    float* partials = (float*)(ws + PART_OFF);
    const unsigned lds_h = (unsigned)(uintptr_t)&h_bf[0];  // wave-relative LDS offset

    v16bf a[32];                                         // register-resident weight frags

    // ================= ENCODER (EMB=512: waves 0..3 active, 1 rowblock each) =======
    if (wave < 4u) {
        const unsigned char* base =
            ws + WENC_OFF + (unsigned long long)((wg * 4u + wave) * 16u) * 1024ull + lane * 32u;
        #pragma unroll
        for (int kb = 0; kb < 16; ++kb)
            a[kb] = *(const v16bf*)(base + (unsigned)kb * 1024u);
    }
    if (tid < 64u) {
        unsigned row = wg * 64u + tid;
        cbuf[tid]  = enc_Wih[row];                 // x_t weight (NF==1)
        owbuf[tid] = enc_bih[row] + enc_bhh[row];  // bias sum
    }
    for (unsigned i = tid; i < EMB; i += TPB) h_bf[i] = 0;  // h0 = 0
    __syncthreads();

    for (int t = 0; t < SEQ; ++t) {
        float xt = x[t];
        unsigned long long* dst = g_henc + (unsigned)(t & 1) * 128u;
        if (wave < 4u) {
            v8f acc0 = {}, acc1 = {};               // two independent WMMA chains
            #pragma unroll
            for (int kb = 0; kb < 16; kb += 2) {
                v16bf be = *(const v16bf*)&h_bf[(unsigned)kb * 32u + hi * 16u];
                v16bf bo = *(const v16bf*)&h_bf[(unsigned)(kb + 1) * 32u + hi * 16u];
                acc0 = __builtin_amdgcn_wmma_f32_16x16x32_bf16(
                           false, a[kb], false, be, (short)0, acc0, false, false);
                acc1 = __builtin_amdgcn_wmma_f32_16x16x32_bf16(
                           false, a[kb + 1], false, bo, (short)0, acc1, false, false);
            }
            v8f acc = acc0 + acc1;
            if ((lane & 15u) == 0u) {               // lanes 0 and 16 hold column 0
                unsigned rlo = hi * 8u;
                float hv[8];
                #pragma unroll
                for (int v = 0; v < 8; ++v) {
                    unsigned rl = wave * 16u + rlo + (unsigned)v;
                    hv[v] = tanhf(acc[v] + xt * cbuf[rl] + owbuf[rl]);
                }
                unsigned long long* dsl = dst + wg * 16u + wave * 4u + hi * 2u;
                __hip_atomic_store(&dsl[0], pack4_bf16(hv[0], hv[1], hv[2], hv[3]),
                                   __ATOMIC_RELAXED, __HIP_MEMORY_SCOPE_AGENT);
                __hip_atomic_store(&dsl[1], pack4_bf16(hv[4], hv[5], hv[6], hv[7]),
                                   __ATOMIC_RELAXED, __HIP_MEMORY_SCOPE_AGENT);
                unsigned long long* esl = g_emb + wg * 32u + wave * 8u + hi * 4u;
                #pragma unroll
                for (int v = 0; v < 4; ++v)
                    __hip_atomic_store(&esl[v], pack2_f32(hv[2 * v], hv[2 * v + 1]),
                                       __ATOMIC_RELAXED, __HIP_MEMORY_SCOPE_AGENT);
                __threadfence();                    // complete before flag publish
            }
        }
        __syncthreads();
        unsigned target = (unsigned)t + 1u;
        if (tid == 0)
            __hip_atomic_store(&flags[wg * 16u], target,
                               __ATOMIC_RELEASE, __HIP_MEMORY_SCOPE_AGENT);
        if (tid < NWG) poll_ge(&flags[tid * 16u], target);
        __syncthreads();
        if (tid < 128u)                             // 512 bf16 = 128 qwords -> LDS
            async_qword_to_lds(lds_h + tid * 8u, dst + tid);
        wait_asynccnt0();
        __syncthreads();
    }

    // ================= DECODER SETUP =================
    {
        const unsigned char* base =
            ws + WDEC_OFF + (unsigned long long)((wg * 8u + wave) * 32u) * 1024ull + lane * 32u;
        #pragma unroll
        for (int kb = 0; kb < 32; ++kb)
            a[kb] = *(const v16bf*)(base + (unsigned)kb * 1024u);
    }
    ((unsigned long long*)ebuf)[tid] =               // emb: 512 f32 = 256 qwords
        __hip_atomic_load(&g_emb[tid], __ATOMIC_RELAXED, __HIP_MEMORY_SCOPE_AGENT);
    __syncthreads();
    if (tid < 128u) {          // dec_c = dec_Wih @ emb + bih + bhh   (once)
        unsigned row = wg * 128u + tid;
        float s = dec_bih[row] + dec_bhh[row];
        const float* wrow = dec_Wih + (unsigned long long)row * EMB;
        for (int j = 0; j < EMB; ++j) s += wrow[j] * ebuf[j];
        cbuf[tid]  = s;
        owbuf[tid] = out_W[row];
    }
    for (unsigned i = tid; i < HID; i += TPB) h_bf[i] = 0;  // h0 = 0
    __syncthreads();

    // ================= DECODER (HID=1024: all 8 waves, 1 rowblock each) ============
    for (int t = 0; t < SEQ; ++t) {
        unsigned long long* dst = g_hdec + (unsigned)(t & 1) * 256u;
        v8f acc0 = {}, acc1 = {};                   // two independent WMMA chains
        #pragma unroll
        for (int kb = 0; kb < 32; kb += 2) {
            v16bf be = *(const v16bf*)&h_bf[(unsigned)kb * 32u + hi * 16u];
            v16bf bo = *(const v16bf*)&h_bf[(unsigned)(kb + 1) * 32u + hi * 16u];
            acc0 = __builtin_amdgcn_wmma_f32_16x16x32_bf16(
                       false, a[kb], false, be, (short)0, acc0, false, false);
            acc1 = __builtin_amdgcn_wmma_f32_16x16x32_bf16(
                       false, a[kb + 1], false, bo, (short)0, acc1, false, false);
        }
        v8f acc = acc0 + acc1;
        if ((lane & 15u) == 0u) {
            unsigned rlo = hi * 8u;
            float hv[8];
            float p = 0.f;
            #pragma unroll
            for (int v = 0; v < 8; ++v) {
                unsigned rl = wave * 16u + rlo + (unsigned)v;
                hv[v] = tanhf(acc[v] + cbuf[rl]);
                p += owbuf[rl] * hv[v];             // fold output projection in
            }
            pbuf[wave * 2u + hi] = p;
            unsigned long long* dsl = dst + wg * 32u + wave * 4u + hi * 2u;
            __hip_atomic_store(&dsl[0], pack4_bf16(hv[0], hv[1], hv[2], hv[3]),
                               __ATOMIC_RELAXED, __HIP_MEMORY_SCOPE_AGENT);
            __hip_atomic_store(&dsl[1], pack4_bf16(hv[4], hv[5], hv[6], hv[7]),
                               __ATOMIC_RELAXED, __HIP_MEMORY_SCOPE_AGENT);
            __threadfence();                        // complete before flag publish
        }
        __syncthreads();
        unsigned target = (unsigned)SEQ + (unsigned)t + 1u;
        if (tid == 0)
            __hip_atomic_store(&flags[wg * 16u], target,
                               __ATOMIC_RELEASE, __HIP_MEMORY_SCOPE_AGENT);
        if (tid == 32u) {       // deterministic in-order partial sum for this WG
            float s = 0.f;
            #pragma unroll
            for (int i2 = 0; i2 < 16; ++i2) s += pbuf[i2];
            partials[(unsigned long long)t * NWG + wg] = s;
        }
        if (tid < NWG) poll_ge(&flags[tid * 16u], target);
        __syncthreads();
        async_qword_to_lds(lds_h + tid * 8u, dst + tid);  // 1024 bf16 = 256 qwords
        wait_asynccnt0();
        __syncthreads();
    }
}

// ---------------- final output: out[t] = sum_wg partials + out_b ----------------
__global__ void out_kernel(const unsigned char* __restrict__ ws,
                           const float* __restrict__ out_b,
                           float* __restrict__ out) {
    unsigned t = blockIdx.x * blockDim.x + threadIdx.x;
    if (t >= SEQ) return;
    const float* partials = (const float*)(ws + PART_OFF);
    float s = out_b[0];
    #pragma unroll
    for (int i = 0; i < NWG; ++i) s += partials[(unsigned long long)t * NWG + i];
    out[t] = s;
}

// ---------------- host entry ----------------
extern "C" void kernel_launch(void* const* d_in, const int* in_sizes, int n_in,
                              void* d_out, int out_size, void* d_ws, size_t ws_size,
                              hipStream_t stream) {
    (void)in_sizes; (void)n_in; (void)out_size; (void)ws_size;
    const float* x       = (const float*)d_in[0];
    const float* enc_Wih = (const float*)d_in[1];
    const float* enc_Whh = (const float*)d_in[2];
    const float* enc_bih = (const float*)d_in[3];
    const float* enc_bhh = (const float*)d_in[4];
    const float* dec_Wih = (const float*)d_in[5];
    const float* dec_Whh = (const float*)d_in[6];
    const float* dec_bih = (const float*)d_in[7];
    const float* dec_bhh = (const float*)d_in[8];
    const float* out_W   = (const float*)d_in[9];
    const float* out_b   = (const float*)d_in[10];
    unsigned char* ws = (unsigned char*)d_ws;
    float* out = (float*)d_out;

    prep_kernel<<<4096, 256, 0, stream>>>(enc_Whh, dec_Whh, ws);
    rnn_kernel<<<NWG, TPB, 0, stream>>>(x, enc_Wih, enc_bih, enc_bhh,
                                        dec_Wih, dec_bih, dec_bhh, out_W, ws);
    out_kernel<<<SEQ / 256, 256, 0, stream>>>(ws, out_b, out);
}
